// MultiHeadSelfAttention_15040975471108
// MI455X (gfx1250) — matrix operand order repaired and checked by reading, not compiled, not run on hardware
//
#include <hip/hip_runtime.h>

// ---------------------------------------------------------------------------
// MI455X / gfx1250 multi-head self-attention with RoPE.
// Matmuls -> V_WMMA_F32_16X16X32_BF16 (wave32); tile movement -> TDM
// (tensor_load_to_lds + s_wait_tensorcnt) when the builtin is available.
// ---------------------------------------------------------------------------

typedef __attribute__((ext_vector_type(16))) __bf16        v16bf;
typedef __attribute__((ext_vector_type(8)))  float         v8f;
typedef __attribute__((ext_vector_type(4)))  unsigned int  u32x4;
typedef __attribute__((ext_vector_type(4)))  int           i32x4;
typedef __attribute__((ext_vector_type(8)))  int           i32x8;

#define S_LEN   4096
#define D_MODEL 1024
#define N_HEADS 16
#define D_HEAD  64
#define LDST    40   // padded LDS row stride (bf16): 80B rows, 16B-aligned

#if __has_builtin(__builtin_amdgcn_tensor_load_to_lds) && \
    __has_builtin(__builtin_amdgcn_s_wait_tensorcnt)
#define USE_TDM 1
#else
#define USE_TDM 0
#endif

union Frag16 { v16bf v; u32x4 q[2]; };

__device__ __forceinline__ v8f v8f_zero() {
  v8f z = {0.f, 0.f, 0.f, 0.f, 0.f, 0.f, 0.f, 0.f};
  return z;
}

#define WMMA_BF16(A, B, C) \
  __builtin_amdgcn_wmma_f32_16x16x32_bf16(false, (A), false, (B), (short)0, (C), false, false)

// A-fragment 16x32 (MxK): lanes 0-15 row M=lane K 0..15;
// lanes 16-31 row M=lane-16 K 16..31. Two b128 loads per lane.
__device__ __forceinline__ v16bf load_a_frag(const __bf16* base, int stride) {
  const int l = threadIdx.x & 31;
  const __bf16* p = base + (l & 15) * stride + ((l >> 4) << 4);
  Frag16 f;
  f.q[0] = *(const u32x4*)(p);
  f.q[1] = *(const u32x4*)(p + 8);
  return f.v;
}

// B-fragment 32x16 (KxN): lanes 0-15 col N=lane hold K 0..15;
// lanes 16-31 col N=lane-16 hold K 16..31.
__device__ __forceinline__ v16bf load_b_frag(const __bf16* base, int stride) {
  const int l = threadIdx.x & 31;
  const __bf16* p = base + (l & 15) * stride + ((l >> 4) << 4);
  Frag16 f;
  f.q[0] = *(const u32x4*)(p);
  f.q[1] = *(const u32x4*)(p + 8);
  return f.v;
}

#if USE_TDM
// Issue one TDM 2D tile load: 64 rows x 32 bf16 cols from a row-major tensor
// with row stride `stride0` elems, into LDS at byte offset lds_addr with
// +16B padding every 64B (-> 40-element LDS row stride).
// D# built per CDNA5 ISA 8.3/8.4 field layout.
__device__ __forceinline__ void tdm_load_tile(const void* gaddr, unsigned lds_addr,
                                              unsigned tdim0, unsigned tdim1,
                                              unsigned stride0) {
  const unsigned long long ga = (unsigned long long)(size_t)gaddr;
  u32x4 g0;
  g0[0] = 1u;                                   // count=1, user descriptor
  g0[1] = lds_addr;                             // lds_addr bytes
  g0[2] = (unsigned)ga;                         // global_addr[31:0]
  g0[3] = (unsigned)(ga >> 32) | (2u << 30);    // global_addr[56:32] | type=2
  i32x8 g1;
  // data_size=2B, pad_enable, pad_interval=16 DWORDs (enc 3), pad_amount=4 DWORDs (enc 3)
  g1[0] = (int)((1u << 16) | (1u << 20) | (3u << 22) | (3u << 25));
  g1[1] = (int)((tdim0 & 0xFFFFu) << 16);               // tensor_dim0[15:0]
  g1[2] = (int)((tdim0 >> 16) | ((tdim1 & 0xFFFFu) << 16)); // dim0 hi | dim1 lo
  g1[3] = (int)((tdim1 >> 16) | (32u << 16));           // dim1 hi | tile_dim0=32
  g1[4] = (int)(64u);                                   // tile_dim1=64, tile_dim2=0
  g1[5] = (int)stride0;                                 // tensor_dim0_stride lo32
  g1[6] = 0;                                            // stride0 hi | dim1_stride lo
  g1[7] = 0;
  i32x4 z4 = {0, 0, 0, 0};
#if defined(__clang_major__) && (__clang_major__ >= 23)
  i32x8 z8 = {0, 0, 0, 0, 0, 0, 0, 0};
  __builtin_amdgcn_tensor_load_to_lds(g0, g1, z4, z4, z8, 0);
#else
  __builtin_amdgcn_tensor_load_to_lds(g0, g1, z4, z4, 0);
#endif
}

__device__ __forceinline__ unsigned lds_byte_off(const void* p) {
  return (unsigned)(unsigned long long)(size_t)p;
}
#endif // USE_TDM

// ---------------------------------------------------------------------------
// fp32 -> bf16 conversion
// ---------------------------------------------------------------------------
__global__ void f32_to_bf16_kernel(const float* __restrict__ in,
                                   __bf16* __restrict__ out, int n) {
  int i = blockIdx.x * blockDim.x + threadIdx.x;
  if (i < n) out[i] = (__bf16)in[i];
}

// ---------------------------------------------------------------------------
// Fused QKV projection: C = X @ W^T for 3 weights sharing the X tile.
// Block = 128 threads (4 waves); tile 64x64x32; 12 WMMAs per k-step.
// ---------------------------------------------------------------------------
__global__ __launch_bounds__(128) void qkv_gemm_kernel(
    const __bf16* __restrict__ X,
    const __bf16* __restrict__ Wq, const __bf16* __restrict__ Wk,
    const __bf16* __restrict__ Wv,
    __bf16* __restrict__ Q, __bf16* __restrict__ K, __bf16* __restrict__ V) {
  const int wave = threadIdx.x >> 5;
  const int lane = threadIdx.x & 31;
  const int rowBase = blockIdx.y * 64;
  const int colBase = blockIdx.x * 64;

  v8f aq[4], ak[4], av[4];
#pragma unroll
  for (int g = 0; g < 4; ++g) { aq[g] = v8f_zero(); ak[g] = v8f_zero(); av[g] = v8f_zero(); }

#if USE_TDM
  __shared__ __bf16 Xs[2][64 * LDST];
  __shared__ __bf16 Qs[2][64 * LDST];
  __shared__ __bf16 Ks[2][64 * LDST];
  __shared__ __bf16 Vs[2][64 * LDST];

  const __bf16* Xrow = X  + (size_t)rowBase * D_MODEL;
  const __bf16* Wqr  = Wq + (size_t)colBase * D_MODEL;
  const __bf16* Wkr  = Wk + (size_t)colBase * D_MODEL;
  const __bf16* Wvr  = Wv + (size_t)colBase * D_MODEL;

  if (wave == 0) {  // TDM ignores EXEC; issue once per block from wave 0
    tdm_load_tile(Xrow, lds_byte_off(&Xs[0][0]), D_MODEL, S_LEN,   D_MODEL);
    tdm_load_tile(Wqr,  lds_byte_off(&Qs[0][0]), D_MODEL, D_MODEL, D_MODEL);
    tdm_load_tile(Wkr,  lds_byte_off(&Ks[0][0]), D_MODEL, D_MODEL, D_MODEL);
    tdm_load_tile(Wvr,  lds_byte_off(&Vs[0][0]), D_MODEL, D_MODEL, D_MODEL);
  }

  for (int k0 = 0; k0 < D_MODEL; k0 += 32) {
    const int cur = (k0 >> 5) & 1;
    if (k0 + 32 < D_MODEL) {
      const int nxt = cur ^ 1;
      if (wave == 0) {
        tdm_load_tile(Xrow + k0 + 32, lds_byte_off(&Xs[nxt][0]), D_MODEL, S_LEN,   D_MODEL);
        tdm_load_tile(Wqr  + k0 + 32, lds_byte_off(&Qs[nxt][0]), D_MODEL, D_MODEL, D_MODEL);
        tdm_load_tile(Wkr  + k0 + 32, lds_byte_off(&Ks[nxt][0]), D_MODEL, D_MODEL, D_MODEL);
        tdm_load_tile(Wvr  + k0 + 32, lds_byte_off(&Vs[nxt][0]), D_MODEL, D_MODEL, D_MODEL);
      }
      __builtin_amdgcn_s_wait_tensorcnt(4);  // current tile's 4 loads done (in-order)
    } else {
      __builtin_amdgcn_s_wait_tensorcnt(0);
    }
    __syncthreads();

    const v16bf af = load_a_frag(&Xs[cur][wave * 16 * LDST], LDST);
    v16bf bq[4], bk[4], bv[4];
#pragma unroll
    for (int g = 0; g < 4; ++g) {
      bq[g] = load_b_frag(&Qs[cur][g * 16 * LDST], LDST);
      bk[g] = load_b_frag(&Ks[cur][g * 16 * LDST], LDST);
      bv[g] = load_b_frag(&Vs[cur][g * 16 * LDST], LDST);
    }
#pragma unroll
    for (int g = 0; g < 4; ++g) {
      aq[g] = WMMA_BF16(af, bq[g], aq[g]);
      ak[g] = WMMA_BF16(af, bk[g], ak[g]);
      av[g] = WMMA_BF16(af, bv[g], av[g]);
    }
    __syncthreads();
  }
#else  // !USE_TDM: synchronous cooperative copies
  __shared__ __bf16 Xs[64 * LDST];
  __shared__ __bf16 Qs[64 * LDST];
  __shared__ __bf16 Ks[64 * LDST];
  __shared__ __bf16 Vs[64 * LDST];

  for (int k0 = 0; k0 < D_MODEL; k0 += 32) {
#pragma unroll
    for (int i = 0; i < 2; ++i) {
      const int chunk = threadIdx.x + i * 128;
      const int r  = chunk >> 2;
      const int c8 = (chunk & 3) * 8;
      const size_t ga = (size_t)(rowBase + r) * D_MODEL + k0 + c8;
      const size_t gb = (size_t)(colBase + r) * D_MODEL + k0 + c8;
      *(u32x4*)(&Xs[r * LDST + c8]) = *(const u32x4*)(&X[ga]);
      *(u32x4*)(&Qs[r * LDST + c8]) = *(const u32x4*)(&Wq[gb]);
      *(u32x4*)(&Ks[r * LDST + c8]) = *(const u32x4*)(&Wk[gb]);
      *(u32x4*)(&Vs[r * LDST + c8]) = *(const u32x4*)(&Wv[gb]);
    }
    if (k0 + 32 < D_MODEL) {
      __builtin_prefetch(&X[(size_t)(rowBase + (threadIdx.x >> 1)) * D_MODEL + k0 + 32], 0, 1);
    }
    __syncthreads();

    const v16bf af = load_a_frag(&Xs[wave * 16 * LDST], LDST);
    v16bf bq[4], bk[4], bv[4];
#pragma unroll
    for (int g = 0; g < 4; ++g) {
      bq[g] = load_b_frag(&Qs[g * 16 * LDST], LDST);
      bk[g] = load_b_frag(&Ks[g * 16 * LDST], LDST);
      bv[g] = load_b_frag(&Vs[g * 16 * LDST], LDST);
    }
#pragma unroll
    for (int g = 0; g < 4; ++g) {
      aq[g] = WMMA_BF16(af, bq[g], aq[g]);
      ak[g] = WMMA_BF16(af, bk[g], ak[g]);
      av[g] = WMMA_BF16(af, bv[g], av[g]);
    }
    __syncthreads();
  }
#endif

  // C/D fragment layout: VGPR v -> row (v + 8*(lane>>4)), col (lane&15)
  const int n = lane & 15, hi = lane >> 4;
#pragma unroll
  for (int g = 0; g < 4; ++g) {
#pragma unroll
    for (int v = 0; v < 8; ++v) {
      const size_t idx = (size_t)(rowBase + wave * 16 + v + hi * 8) * D_MODEL
                       + colBase + g * 16 + n;
      Q[idx] = (__bf16)aq[g][v];
      K[idx] = (__bf16)ak[g][v];
      V[idx] = (__bf16)av[g][v];
    }
  }
}

// ---------------------------------------------------------------------------
// RoPE in-place on [s][h*d_k] bf16 buffer; one thread per (even,odd) pair.
// ---------------------------------------------------------------------------
__global__ void rope_kernel(__bf16* __restrict__ T) {
  const int idx = blockIdx.x * blockDim.x + threadIdx.x;   // < 4096*512
  const int pos  = idx >> 9;
  const int p    = idx & 511;
  const int head = p >> 5;
  const int i    = p & 31;
  const float inv = __powf(10000.0f, -(2.0f * (float)i) / 64.0f);
  const float ang = (float)pos * inv;
  float s, c;
  __sincosf(ang, &s, &c);
  const size_t base = (size_t)pos * D_MODEL + head * D_HEAD + 2 * i;
  const float t1 = (float)T[base];
  const float t2 = (float)T[base + 1];
  T[base]     = (__bf16)(t1 * c - t2 * s);
  T[base + 1] = (__bf16)(t1 * s + t2 * c);
}

// ---------------------------------------------------------------------------
// V[s][c] -> Vt[c][s]  (c = h*64 + dk), so P@V B-fragments are contiguous.
// ---------------------------------------------------------------------------
__global__ void transpose_v_kernel(const __bf16* __restrict__ V,
                                   __bf16* __restrict__ Vt) {
  const int idx = blockIdx.x * blockDim.x + threadIdx.x;   // < 4096*1024
  const int s = idx >> 10;
  const int c = idx & 1023;
  Vt[(size_t)c * S_LEN + s] = V[idx];
}

// ---------------------------------------------------------------------------
// Causal flash attention. grid = (s/64, heads); block = 128 (4 waves).
// Wave owns 16 q-rows; loops over 32-key tiles up to its causal limit.
// ---------------------------------------------------------------------------
__global__ __launch_bounds__(128) void flash_attn_kernel(
    const __bf16* __restrict__ Q, const __bf16* __restrict__ K,
    const __bf16* __restrict__ Vt, __bf16* __restrict__ O) {
  __shared__ __bf16 pLds[4][16 * LDST];   // per-wave private P staging

  const int wave = threadIdx.x >> 5;
  const int lane = threadIdx.x & 31;
  const int n  = lane & 15;
  const int hi = lane >> 4;
  const int h  = blockIdx.y;
  const int qb = blockIdx.x * 64 + wave * 16;

  // Q A-fragments (dk 0..31 and 32..63), loaded once.
  v16bf qf0, qf1;
  {
    const __bf16* qp = Q + (size_t)(qb + n) * D_MODEL + h * D_HEAD + hi * 16;
    Frag16 f0, f1;
    f0.q[0] = *(const u32x4*)(qp);
    f0.q[1] = *(const u32x4*)(qp + 8);
    f1.q[0] = *(const u32x4*)(qp + 32);
    f1.q[1] = *(const u32x4*)(qp + 40);
    qf0 = f0.v; qf1 = f1.v;
  }

  v8f o0 = v8f_zero(), o1 = v8f_zero(), o2 = v8f_zero(), o3 = v8f_zero();
  float mrow[8], lrow[8];
#pragma unroll
  for (int v = 0; v < 8; ++v) { mrow[v] = -3.0e38f; lrow[v] = 0.0f; }
  const float scale = 0.125f;   // 1/sqrt(64)
  const int qhi = qb + 15;

  for (int kt = 0; kt <= qhi; kt += 32) {
    // Scores for two 16-key groups; batch 4 fragment loads then 4 WMMAs.
    v8f s0, s1;
    {
      const __bf16* kp0 = K + (size_t)(kt + n) * D_MODEL + h * D_HEAD + hi * 16;
      const __bf16* kp1 = kp0 + (size_t)16 * D_MODEL;
      Frag16 a0, a1, c0, c1;
      a0.q[0] = *(const u32x4*)(kp0);       a0.q[1] = *(const u32x4*)(kp0 + 8);
      a1.q[0] = *(const u32x4*)(kp0 + 32);  a1.q[1] = *(const u32x4*)(kp0 + 40);
      c0.q[0] = *(const u32x4*)(kp1);       c0.q[1] = *(const u32x4*)(kp1 + 8);
      c1.q[0] = *(const u32x4*)(kp1 + 32);  c1.q[1] = *(const u32x4*)(kp1 + 40);
      v8f z0 = v8f_zero(), z1 = v8f_zero();
      z0 = WMMA_BF16(qf0, a0.v, z0);
      z1 = WMMA_BF16(qf0, c0.v, z1);
      s0 = WMMA_BF16(qf1, a1.v, z0);
      s1 = WMMA_BF16(qf1, c1.v, z1);
    }

    // Causal mask + online softmax per fragment row (reductions across the
    // 16-lane half via shfl_xor 1/2/4/8).
#pragma unroll
    for (int v = 0; v < 8; ++v) {
      const int qrow = qb + v + hi * 8;
      float a = s0[v] * scale; if (kt + n > qrow)      a = -1.0e30f;
      float b = s1[v] * scale; if (kt + 16 + n > qrow) b = -1.0e30f;
      float mx = fmaxf(a, b);
      mx = fmaxf(mx, __shfl_xor(mx, 1, 32));
      mx = fmaxf(mx, __shfl_xor(mx, 2, 32));
      mx = fmaxf(mx, __shfl_xor(mx, 4, 32));
      mx = fmaxf(mx, __shfl_xor(mx, 8, 32));
      const float mnew = fmaxf(mrow[v], mx);
      const float corr = __expf(mrow[v] - mnew);
      const float pa = __expf(a - mnew);
      const float pb = __expf(b - mnew);
      float rs = pa + pb;
      rs += __shfl_xor(rs, 1, 32);
      rs += __shfl_xor(rs, 2, 32);
      rs += __shfl_xor(rs, 4, 32);
      rs += __shfl_xor(rs, 8, 32);
      lrow[v] = lrow[v] * corr + rs;
      mrow[v] = mnew;
      o0[v] *= corr; o1[v] *= corr; o2[v] *= corr; o3[v] *= corr;
      // Stage P (C-frag layout -> row-major bf16 16x32 tile).
      const int m = v + hi * 8;
      pLds[wave][m * LDST + n]      = (__bf16)pa;
      pLds[wave][m * LDST + 16 + n] = (__bf16)pb;
    }

    // Reshape P through LDS into an A-fragment; accumulate O += P @ V.
    const v16bf pf = load_a_frag(&pLds[wave][0], LDST);
    {
      const __bf16* vp = Vt + (size_t)(h * D_HEAD + n) * S_LEN + kt + hi * 16;
      Frag16 vf0, vf1, vf2, vf3;
      vf0.q[0] = *(const u32x4*)(vp);  vf0.q[1] = *(const u32x4*)(vp + 8);
      vp += (size_t)16 * S_LEN;
      vf1.q[0] = *(const u32x4*)(vp);  vf1.q[1] = *(const u32x4*)(vp + 8);
      vp += (size_t)16 * S_LEN;
      vf2.q[0] = *(const u32x4*)(vp);  vf2.q[1] = *(const u32x4*)(vp + 8);
      vp += (size_t)16 * S_LEN;
      vf3.q[0] = *(const u32x4*)(vp);  vf3.q[1] = *(const u32x4*)(vp + 8);
      o0 = WMMA_BF16(pf, vf0.v, o0);
      o1 = WMMA_BF16(pf, vf1.v, o1);
      o2 = WMMA_BF16(pf, vf2.v, o2);
      o3 = WMMA_BF16(pf, vf3.v, o3);
    }
  }

  // Normalize and write attn output (bf16 [s][h*dk]).
#pragma unroll
  for (int v = 0; v < 8; ++v) {
    const size_t r = (size_t)(qb + v + hi * 8) * D_MODEL + h * D_HEAD + n;
    const float inv = 1.0f / lrow[v];
    O[r]      = (__bf16)(o0[v] * inv);
    O[r + 16] = (__bf16)(o1[v] * inv);
    O[r + 32] = (__bf16)(o2[v] * inv);
    O[r + 48] = (__bf16)(o3[v] * inv);
  }
}

// ---------------------------------------------------------------------------
// Generic NT GEMM: C[M,N] = A[M,K] @ B[N,K]^T, fp32 output (final projection).
// ---------------------------------------------------------------------------
__global__ __launch_bounds__(128) void gemm_nt_f32out_kernel(
    const __bf16* __restrict__ A, const __bf16* __restrict__ B,
    float* __restrict__ C, int M, int N, int Kd) {
  const int wave = threadIdx.x >> 5;
  const int lane = threadIdx.x & 31;
  const int rowBase = blockIdx.y * 64;
  const int colBase = blockIdx.x * 64;

  v8f acc[4];
#pragma unroll
  for (int g = 0; g < 4; ++g) acc[g] = v8f_zero();

#if USE_TDM
  __shared__ __bf16 As[2][64 * LDST];
  __shared__ __bf16 Bs[2][64 * LDST];
  const __bf16* Arow = A + (size_t)rowBase * Kd;
  const __bf16* Brow = B + (size_t)colBase * Kd;

  if (wave == 0) {
    tdm_load_tile(Arow, lds_byte_off(&As[0][0]), Kd, M, Kd);
    tdm_load_tile(Brow, lds_byte_off(&Bs[0][0]), Kd, N, Kd);
  }
  for (int k0 = 0; k0 < Kd; k0 += 32) {
    const int cur = (k0 >> 5) & 1;
    if (k0 + 32 < Kd) {
      const int nxt = cur ^ 1;
      if (wave == 0) {
        tdm_load_tile(Arow + k0 + 32, lds_byte_off(&As[nxt][0]), Kd, M, Kd);
        tdm_load_tile(Brow + k0 + 32, lds_byte_off(&Bs[nxt][0]), Kd, N, Kd);
      }
      __builtin_amdgcn_s_wait_tensorcnt(2);
    } else {
      __builtin_amdgcn_s_wait_tensorcnt(0);
    }
    __syncthreads();

    const v16bf af = load_a_frag(&As[cur][wave * 16 * LDST], LDST);
    v16bf bf[4];
#pragma unroll
    for (int g = 0; g < 4; ++g) bf[g] = load_b_frag(&Bs[cur][g * 16 * LDST], LDST);
#pragma unroll
    for (int g = 0; g < 4; ++g) acc[g] = WMMA_BF16(af, bf[g], acc[g]);
    __syncthreads();
  }
#else
  __shared__ __bf16 As[64 * LDST];
  __shared__ __bf16 Bs[64 * LDST];

  for (int k0 = 0; k0 < Kd; k0 += 32) {
#pragma unroll
    for (int i = 0; i < 2; ++i) {
      const int chunk = threadIdx.x + i * 128;
      const int r  = chunk >> 2;
      const int c8 = (chunk & 3) * 8;
      *(u32x4*)(&As[r * LDST + c8]) =
          *(const u32x4*)(&A[(size_t)(rowBase + r) * Kd + k0 + c8]);
      *(u32x4*)(&Bs[r * LDST + c8]) =
          *(const u32x4*)(&B[(size_t)(colBase + r) * Kd + k0 + c8]);
    }
    if (k0 + 32 < Kd) {
      __builtin_prefetch(&A[(size_t)(rowBase + (threadIdx.x >> 1)) * Kd + k0 + 32], 0, 1);
    }
    __syncthreads();

    const v16bf af = load_a_frag(&As[wave * 16 * LDST], LDST);
    v16bf bf[4];
#pragma unroll
    for (int g = 0; g < 4; ++g) bf[g] = load_b_frag(&Bs[g * 16 * LDST], LDST);
#pragma unroll
    for (int g = 0; g < 4; ++g) acc[g] = WMMA_BF16(af, bf[g], acc[g]);
    __syncthreads();
  }
#endif

  const int n = lane & 15, hi = lane >> 4;
#pragma unroll
  for (int g = 0; g < 4; ++g) {
#pragma unroll
    for (int v = 0; v < 8; ++v) {
      C[(size_t)(rowBase + wave * 16 + v + hi * 8) * N + colBase + g * 16 + n] =
          acc[g][v];
    }
  }
}

// ---------------------------------------------------------------------------
// Host-side launch sequence (graph-capture safe: kernels only).
// ---------------------------------------------------------------------------
extern "C" void kernel_launch(void* const* d_in, const int* in_sizes, int n_in,
                              void* d_out, int out_size, void* d_ws, size_t ws_size,
                              hipStream_t stream) {
  (void)in_sizes; (void)n_in; (void)out_size; (void)ws_size;
  const float* x   = (const float*)d_in[0];
  const float* w_q = (const float*)d_in[1];
  const float* w_k = (const float*)d_in[2];
  const float* w_v = (const float*)d_in[3];
  const float* w_o = (const float*)d_in[4];
  float* out = (float*)d_out;

  const size_t SD = (size_t)S_LEN * D_MODEL;   // 4M elems
  const size_t DD = (size_t)D_MODEL * D_MODEL; // 1M elems

  __bf16* xb  = (__bf16*)d_ws;        // [s][d]
  __bf16* wqb = xb  + SD;
  __bf16* wkb = wqb + DD;
  __bf16* wvb = wkb + DD;
  __bf16* wob = wvb + DD;
  __bf16* Qb  = wob + DD;             // [s][d]
  __bf16* Kb  = Qb  + SD;             // [s][d]
  __bf16* Vb  = Kb  + SD;             // [s][d]
  __bf16* Vtb = Vb  + SD;             // [d][s]
  __bf16* attnb = Vb;                 // reuse V after transpose

  // 1) fp32 -> bf16
  f32_to_bf16_kernel<<<(int)(SD / 256), 256, 0, stream>>>(x, xb, (int)SD);
  f32_to_bf16_kernel<<<(int)(DD / 256), 256, 0, stream>>>(w_q, wqb, (int)DD);
  f32_to_bf16_kernel<<<(int)(DD / 256), 256, 0, stream>>>(w_k, wkb, (int)DD);
  f32_to_bf16_kernel<<<(int)(DD / 256), 256, 0, stream>>>(w_v, wvb, (int)DD);
  f32_to_bf16_kernel<<<(int)(DD / 256), 256, 0, stream>>>(w_o, wob, (int)DD);

  // 2) fused QKV projection (x tile reused 3x, TDM double-buffered)
  qkv_gemm_kernel<<<dim3(D_MODEL / 64, S_LEN / 64), 128, 0, stream>>>(
      xb, wqb, wkb, wvb, Qb, Kb, Vb);

  // 3) RoPE on Q and K
  rope_kernel<<<(S_LEN * (D_MODEL / 2)) / 256, 256, 0, stream>>>(Qb);
  rope_kernel<<<(S_LEN * (D_MODEL / 2)) / 256, 256, 0, stream>>>(Kb);

  // 4) V -> V^T
  transpose_v_kernel<<<(int)(SD / 256), 256, 0, stream>>>(Vb, Vtb);

  // 5) causal flash attention
  flash_attn_kernel<<<dim3(S_LEN / 64, N_HEADS), 128, 0, stream>>>(
      Qb, Kb, Vtb, attnb);

  // 6) output projection -> fp32 d_out
  gemm_nt_f32out_kernel<<<dim3(D_MODEL / 64, S_LEN / 64), 128, 0, stream>>>(
      attnb, wob, out, S_LEN, D_MODEL, D_MODEL);
}
